// GEPSConv1D_64098091925939
// MI455X (gfx1250) — matrix operand order repaired and checked by reading, not compiled, not run on hardware
//
#include <hip/hip_runtime.h>
#include <hip/hip_bf16.h>

typedef __attribute__((ext_vector_type(16))) _Float16 v16h;
typedef __attribute__((ext_vector_type(8)))  float    v8f;

#define B_   32
#define CIN  128
#define COUT 128
#define LEN  4096
#define KW   3
#define CKTOT (CIN * KW)      // 384
#define NCHUNK (CKTOT / 32)   // 12
#define HSTRIPE 128           // h per block (8 waves x 16)
#define XROWS (HSTRIPE + 2)   // 130: halo row on each side (circular)
#define XCOLS 144             // 128 i + pad -> 288B row stride (32B aligned)

// Grid: (32 h-blocks, 8 o-tiles, 32 batches). Block: 256 threads = 8 wave32.
// Each wave computes one 16(o) x 16(h) D tile via 12 fully-unrolled WMMAs,
// fed entirely from LDS (input tile staged once, f32->f16 on store).
__global__ __launch_bounds__(256)
void geps_conv1d_wmma_kernel(const float* __restrict__ input,     // (B,CIN,LEN)
                             const float* __restrict__ codes,     // (B,2,2)
                             const float* __restrict__ weight,    // (COUT,CIN,KW)
                             const float* __restrict__ Ahyp,      // (CIN,2,KW)
                             const float* __restrict__ Bhyp,      // (COUT,2,KW)
                             const float* __restrict__ bias,      // (COUT)
                             const float* __restrict__ bias_ctx,  // (2,COUT)
                             float* __restrict__ out)             // (B,COUT,LEN)
{
    // Transposed input tile: xt[l_local][i], f16. Row stride 288B (32B aligned).
    __shared__ __align__(32) _Float16 xt[XROWS][XCOLS];          // ~37.4 KB
    __shared__ __align__(32) _Float16 lds_a[NCHUNK][32][16];     // 12 KB
    __shared__ float lds_bias[16];

    const int tid = threadIdx.x;
    const int h0  = blockIdx.x * HSTRIPE;
    const int m0  = blockIdx.y;   // o-tile (16 outs)
    const int b   = blockIdx.z;
    const float* inb = input + (size_t)b * CIN * LEN;

    // ---- Phase 0a: stage body rows [h0, h0+128) -- never wraps. ----
    // g -> (i = g>>5, quad q = g&31): float4 of 4 consecutive l, coalesced.
    #pragma unroll 1
    for (int g = tid; g < CIN * (HSTRIPE / 4); g += 256) {
        const int i = g >> 5;
        const int q = g & 31;
        const float4 v = *(const float4*)(inb + (size_t)i * LEN + h0 + q * 4);
        const int r = 1 + q * 4;   // xt row 0 is l = h0-1
        xt[r + 0][i] = (_Float16)v.x;
        xt[r + 1][i] = (_Float16)v.y;
        xt[r + 2][i] = (_Float16)v.z;
        xt[r + 3][i] = (_Float16)v.w;
    }
    // ---- Phase 0b: two halo rows (circular wrap). ----
    if (tid < 2 * CIN) {
        const int r = tid >> 7;            // 0 or 1
        const int i = tid & 127;
        const int rl = r ? (XROWS - 1) : 0;
        const int l  = (h0 - 1 + rl + LEN) & (LEN - 1);
        xt[rl][i] = (_Float16)inb[(size_t)i * LEN + l];
    }

    // ---- Phase 1: combined weights in WMMA A-fragment layout in LDS ----
    // ck = k*128 + i. A-frag (16-bit 16x32): lane L row M=L%16;
    // element p: k_local = (L>=16 ? 16 : 0) + p (contiguous order chosen here; B matches).
    // Chunk c: k = c>>2 (constant), i = (c&3)*32 + k_local.
    #pragma unroll 1
    for (int e = tid; e < NCHUNK * 32 * 16; e += 256) {
        const int p    = e & 15;
        const int lane = (e >> 4) & 31;
        const int c    = e >> 9;
        const int row  = lane & 15;
        const int o    = m0 * 16 + row;
        const int kl   = ((lane >> 4) << 4) + p;
        const int k    = c >> 2;
        const int i    = ((c & 3) << 5) + kl;

        float v = weight[(o * CIN + i) * KW + k];
        #pragma unroll
        for (int cc = 0; cc < 2; ++cc)
            #pragma unroll
            for (int rr = 0; rr < 2; ++rr)
                v += Ahyp[(i * 2 + cc) * KW + k]
                   * codes[b * 4 + cc * 2 + rr]
                   * Bhyp[(o * 2 + rr) * KW + k];
        lds_a[c][lane][p] = (_Float16)v;
    }
    if (tid < 16) {
        const int o = m0 * 16 + tid;
        float cb = bias[o];
        #pragma unroll
        for (int r = 0; r < 2; ++r)
            cb += codes[b * 4 + r * 2 + r] * bias_ctx[r * COUT + o];
        lds_bias[tid] = cb;
    }
    __syncthreads();

    // ---- Phase 2: 12 WMMAs per wave, all operands from LDS ----
    const int lane  = tid & 31;
    const int wave  = tid >> 5;
    const int hlane = lane & 15;          // N coordinate
    const int khalf = lane >> 4;          // B frag: K = p + 16*khalf

    v8f acc = {};
    #pragma unroll
    for (int c = 0; c < NCHUNK; ++c) {
        const v16h av = *(const v16h*)(&lds_a[c][lane][0]);
        const int k   = c >> 2;                           // conv tap
        const int ib0 = ((c & 3) << 5) + (khalf << 4);    // i of element p=0
        // xt row for output h = h0 + wave*16 + hlane, tap k:
        // l = h + k - 1  ->  row = wave*16 + hlane + k   (row 0 == h0-1)
        const int rl  = wave * 16 + hlane + k;
        const v16h bv = *(const v16h*)(&xt[rl][ib0]);     // 32B contiguous

        acc = __builtin_amdgcn_wmma_f32_16x16x32_f16(
            /*neg_a=*/false, av, /*neg_b=*/false, bv,
            /*c_mod=*/(short)0, acc,
            /*reuse_a=*/false, /*reuse_b=*/false);
    }

    // ---- Epilogue: D layout (VGPR j: M=j lanes 0-15, M=j+8 lanes 16-31) ----
    const int h = h0 + wave * 16 + hlane;
    float* outb = out + ((size_t)b * COUT + m0 * 16) * LEN;
    #pragma unroll
    for (int j = 0; j < 8; ++j) {
        const int ro = j + (khalf << 3);
        outb[(size_t)ro * LEN + h] = acc[j] + lds_bias[ro];
    }
}

extern "C" void kernel_launch(void* const* d_in, const int* in_sizes, int n_in,
                              void* d_out, int out_size, void* d_ws, size_t ws_size,
                              hipStream_t stream) {
    (void)in_sizes; (void)n_in; (void)out_size; (void)d_ws; (void)ws_size;
    const float* input    = (const float*)d_in[0];
    const float* codes    = (const float*)d_in[1];
    const float* weight   = (const float*)d_in[2];
    const float* Ahyp     = (const float*)d_in[3];
    const float* Bhyp     = (const float*)d_in[4];
    const float* bias     = (const float*)d_in[5];
    const float* bias_ctx = (const float*)d_in[6];
    float* out = (float*)d_out;

    dim3 grid(LEN / HSTRIPE, COUT / 16, B_);  // (32, 8, 32)
    dim3 block(256);
    geps_conv1d_wmma_kernel<<<grid, block, 0, stream>>>(
        input, codes, weight, Ahyp, Bhyp, bias, bias_ctx, out);
}
